// UnderlyingTwoDimensionalGRU_28209345200592
// MI455X (gfx1250) — compile-verified, matrix operand order checked by reading, not run on hardware
//
#include <hip/hip_runtime.h>

// ---------------------------------------------------------------------------
// CDNA5 (gfx1250) 2-D GRU wavefront, bf16 WMMA (v_wmma_f32_16x16x32_bf16),
// fp32 accumulation. wave32, 8 waves / 256-thread workgroup, 1 WGP each.
// ---------------------------------------------------------------------------

typedef __bf16 bf16;
typedef __bf16 v16bf __attribute__((ext_vector_type(16)));
typedef float  v8f   __attribute__((ext_vector_type(8)));

#define WMMA_BF16(a, b, c) \
  __builtin_amdgcn_wmma_f32_16x16x32_bf16(false, (a), false, (b), (short)0, (c), false, false)

// ---- WMMA 16-bit fragment layout helpers (16x32 A-tile / 32x16 B-tile) ----
// Per ISA: lanes 0-15 hold rows M (A) / cols N (B) with K in {0..7,16..23},
// lanes 16-31 hold K in {8..15,24..31}; VGPR r holds a K pair.
__device__ __forceinline__ int frag_k(int lane, int v) {
  int r = v >> 1, lo = v & 1;
  int base = (lane & 16) ? 8 : 0;
  return (r < 4) ? (base + 2 * r + lo) : (16 + base + 2 * (r - 4) + lo);
}

// Offset (in halfwords) of element (M, Kabs) inside an A-buffer laid out as
// [k_tile][m_tile][lane][16 bf16]; tile = 16 rows x 32 K = 512 halves.
__device__ __forceinline__ int a_off(int M, int Kabs) {
  int kt = Kabs >> 5, kl = Kabs & 31;
  int m = M >> 4, Ml = M & 15;
  int hi = (kl >> 3) & 1;            // second lane-half?
  int lane = Ml + hi * 16;
  int kk = kl - hi * 8;              // now in {0..7} or {16..23}
  int r = (kk < 8) ? (kk >> 1) : (4 + ((kk - 16) >> 1));
  int v = r * 2 + (kk & 1);
  return ((kt * 2 + m) * 32 + lane) * 16 + v;
}

// ---------------- workspace layout (bytes) ----------------
static const size_t E_PACK_OFF = 0;                       // 1024 cells * 8192 bf16
static const size_t E_PACK_SZ  = (size_t)1024 * 8192 * 2; // 16,777,216
static const size_t WRZ_OFF    = E_PACK_OFF + E_PACK_SZ;  // 128nt*40kt*512 bf16
static const size_t WRZ_SZ     = (size_t)128 * 40 * 512 * 2;
static const size_t BNI_OFF    = WRZ_OFF + WRZ_SZ;        // 64nt*8kt*512
static const size_t BNI_SZ     = (size_t)64 * 8 * 512 * 2;
static const size_t BNH_OFF    = BNI_OFF + BNI_SZ;        // 64nt*32kt*512
static const size_t BNH_SZ     = (size_t)64 * 32 * 512 * 2;
static const size_t WC_OFF     = BNH_OFF + BNH_SZ;        // 32nt*40kt*512
static const size_t WC_SZ      = (size_t)32 * 40 * 512 * 2;
static const size_t EMB_OFF    = WC_OFF + WC_SZ;          // 16nt*8kt*512
static const size_t EMB_SZ     = (size_t)16 * 8 * 512 * 2;
static const size_t FLAG_OFF   = EMB_OFF + EMB_SZ;        // 1024 ints

static const int WF_SMEM_BYTES = 40960 * 2 + 32768 * 2 + 6208 * 4; // 172,288

// ---------------------------------------------------------------------------
// Pack an (N x K) fp32 weight (optionally split at kSplit between two sources,
// fused-K GEMM) into bf16 WMMA B-fragment tiles: [nt][kt][lane][16].
// ---------------------------------------------------------------------------
__global__ void pack_b_kernel(bf16* __restrict__ dst,
                              const float* __restrict__ srcA, int ldA, int rowOffA,
                              const float* __restrict__ srcB, int ldB, int rowOffB,
                              int kSplit, int NKT, int total) {
  int idx = blockIdx.x * 256 + threadIdx.x;
  if (idx >= total) return;
  int v    = idx & 15;
  int lane = (idx >> 4) & 31;
  int kt   = (idx >> 9) % NKT;
  int nt   = idx / (NKT << 9);
  int n = nt * 16 + (lane & 15);
  int k = kt * 32 + frag_k(lane, v);
  float val = (k < kSplit) ? srcA[(size_t)(rowOffA + n) * ldA + k]
                           : srcB[(size_t)(rowOffB + n) * ldB + (k - kSplit)];
  dst[idx] = (bf16)val;
}

// ---------------------------------------------------------------------------
// Phase 1: e = x @ emb_W^T + emb_b, stored as bf16 A-fragment tiles per cell.
// One workgroup per cell; M=32 (batch), N=256, K=256 via WMMA.
// ---------------------------------------------------------------------------
__global__ __launch_bounds__(256) void embed_kernel(
    const float* __restrict__ x, const float* __restrict__ emb_b,
    const bf16* __restrict__ embB, bf16* __restrict__ ePack) {
  __shared__ bf16 Xb[8 * 1024];   // 8 k-tiles x (2 m x 32 lanes x 16)
  const int cell = blockIdx.x;
  const int ci = cell >> 5, cj = cell & 31;
  const int tid = threadIdx.x;

  for (int t = tid; t < 32 * 256; t += 256) {
    int row = t >> 8, d = t & 255;
    float fv = x[(((size_t)row * 32 + ci) * 32 + cj) * 256 + d];
    Xb[a_off(row, d)] = (bf16)fv;
  }
  __syncthreads();

  const int w = tid >> 5, lane = tid & 31;
  v8f acc[2][2] = {};
  for (int kt = 0; kt < 8; ++kt) {
    v16bf a0 = *(const v16bf*)(Xb + kt * 1024 + lane * 16);
    v16bf a1 = *(const v16bf*)(Xb + kt * 1024 + 512 + lane * 16);
#pragma unroll
    for (int nn = 0; nn < 2; ++nn) {
      int nt = w * 2 + nn;
      v16bf b = *(const v16bf*)(embB + ((size_t)(nt * 8 + kt) * 32 + lane) * 16);
      acc[0][nn] = WMMA_BF16(a0, b, acc[0][nn]);
      acc[1][nn] = WMMA_BF16(a1, b, acc[1][nn]);
    }
  }

  bf16* edst = ePack + (size_t)cell * 8192;
#pragma unroll
  for (int m = 0; m < 2; ++m)
#pragma unroll
    for (int nn = 0; nn < 2; ++nn)
#pragma unroll
      for (int v = 0; v < 8; ++v) {
        int row = m * 16 + v + ((lane & 16) ? 8 : 0);
        int col = (w * 2 + nn) * 16 + (lane & 15);
        float val = acc[m][nn][v] + emb_b[col];
        edst[a_off(row, col)] = (bf16)val;
      }
}

// ---------------------------------------------------------------------------
// Phase 2: wavefront. One persistent workgroup per grid column j (32 total),
// sequential over rows i; cross-column dependency via device-scope flags.
// A-buffer K layout: [0,256)=e, [256,768)=h_above, [768,1280)=h_left.
// ---------------------------------------------------------------------------
__global__ __launch_bounds__(256) void wavefront_kernel(
    const bf16* __restrict__ ePack, const bf16* __restrict__ Wrz,
    const bf16* __restrict__ Bni, const bf16* __restrict__ Bnh,
    const bf16* __restrict__ Wc,
    const float* __restrict__ b_ih, const float* __restrict__ b_hh,
    const float* __restrict__ comp_b, const float* __restrict__ res_b,
    const float* __restrict__ ln_g, const float* __restrict__ ln_b,
    float* __restrict__ out, int* __restrict__ flags) {
  extern __shared__ char smem[];
  bf16* Abuf = (bf16*)smem;                        // 40 k-tiles = 40960 halves
  bf16* Hbuf = (bf16*)(smem + 40960 * 2);          // 32 k-tiles = 32768 halves
  float* fb    = (float*)(smem + 40960 * 2 + 32768 * 2);
  float* biasR  = fb;            // 1024
  float* biasZ  = fb + 1024;
  float* biasNi = fb + 2048;
  float* biasNh = fb + 3072;
  float* biasC  = fb + 4096;     // 512 (comp_b + res_b)
  float* lng    = fb + 4608;     // 512
  float* lnb    = fb + 5120;     // 512
  float* redS   = fb + 5632;     // 8 waves x 32 rows
  float* redQ   = fb + 5888;
  float* stat   = fb + 6144;     // 32 x {mu, rstd}

  const int tid = threadIdx.x;
  const int j   = blockIdx.x;
  const int w = tid >> 5, lane = tid & 31;

  for (int t = tid; t < 1024; t += 256) {
    biasR[t]  = b_ih[t] + b_hh[t];
    biasZ[t]  = b_ih[1024 + t] + b_hh[1024 + t];
    biasNi[t] = b_ih[2048 + t];
    biasNh[t] = b_hh[2048 + t];
  }
  for (int t = tid; t < 512; t += 256) {
    biasC[t] = comp_b[t] + res_b[t];
    lng[t] = ln_g[t];
    lnb[t] = ln_b[t];
  }

#pragma unroll 1
  for (int i = 0; i < 32; ++i) {
    // ---- wait for left neighbor cell (i, j-1) ----
    if (j > 0 && tid == 0) {
      while (__hip_atomic_load(flags + (i * 32 + j - 1), __ATOMIC_ACQUIRE,
                               __HIP_MEMORY_SCOPE_AGENT) == 0) { }
    }
    __syncthreads();

    // ---- stage e fragments (k-tiles 0..7) ----
    {
      const uint4* src = (const uint4*)(ePack + (size_t)(i * 32 + j) * 8192);
      uint4* dst = (uint4*)Abuf;
      for (int t = tid; t < 1024; t += 256) dst[t] = src[t];
    }
    // ---- stage h_left (k-tiles 24..39) from neighbor's d_out slice ----
    if (j > 0) {
      for (int t = tid; t < 32 * 512; t += 256) {
        int row = t >> 9, col = t & 511;
        float hv = out[(((size_t)row * 32 + i) * 32 + (j - 1)) * 512 + col];
        Abuf[a_off(row, 768 + col)] = (bf16)hv;
      }
    } else {
      uint4 z4 = {0u, 0u, 0u, 0u};
      uint4* dst = (uint4*)(Abuf + 24 * 1024);
      for (int t = tid; t < 2048; t += 256) dst[t] = z4;
    }
    if (i == 0) {  // zero h_above (k-tiles 8..23)
      uint4 z4 = {0u, 0u, 0u, 0u};
      uint4* dst = (uint4*)(Abuf + 8 * 1024);
      for (int t = tid; t < 2048; t += 256) dst[t] = z4;
    }
    __syncthreads();

    // ---- gates: fused [gi|gh] GEMM for r/z, split accumulators for n ----
#pragma unroll 1
    for (int g = 0; g < 8; ++g) {
      const int cb  = w * 128 + g * 16;   // this wave+group's 16 h-columns
      const int ntr = cb >> 4;            // n-tile index within a gate block
      v8f ar[2] = {}, az[2] = {};
#pragma unroll 4
      for (int kt = 0; kt < 40; ++kt) {
        v16bf a0 = *(const v16bf*)(Abuf + kt * 1024 + lane * 16);
        v16bf a1 = *(const v16bf*)(Abuf + kt * 1024 + 512 + lane * 16);
        v16bf br = *(const v16bf*)(Wrz + ((size_t)(ntr * 40 + kt) * 32 + lane) * 16);
        v16bf bz = *(const v16bf*)(Wrz + ((size_t)((64 + ntr) * 40 + kt) * 32 + lane) * 16);
        ar[0] = WMMA_BF16(a0, br, ar[0]);
        ar[1] = WMMA_BF16(a1, br, ar[1]);
        az[0] = WMMA_BF16(a0, bz, az[0]);
        az[1] = WMMA_BF16(a1, bz, az[1]);
      }
      v8f ni[2] = {}, nh[2] = {};
#pragma unroll 4
      for (int kt = 0; kt < 8; ++kt) {    // inn = e @ Wn_ih^T  (K = 256)
        v16bf a0 = *(const v16bf*)(Abuf + kt * 1024 + lane * 16);
        v16bf a1 = *(const v16bf*)(Abuf + kt * 1024 + 512 + lane * 16);
        v16bf b  = *(const v16bf*)(Bni + ((size_t)(ntr * 8 + kt) * 32 + lane) * 16);
        ni[0] = WMMA_BF16(a0, b, ni[0]);
        ni[1] = WMMA_BF16(a1, b, ni[1]);
      }
#pragma unroll 4
      for (int kt = 8; kt < 40; ++kt) {   // hn = h_prev @ Wn_hh^T (K = 1024)
        v16bf a0 = *(const v16bf*)(Abuf + kt * 1024 + lane * 16);
        v16bf a1 = *(const v16bf*)(Abuf + kt * 1024 + 512 + lane * 16);
        v16bf b  = *(const v16bf*)(Bnh + ((size_t)(ntr * 32 + (kt - 8)) * 32 + lane) * 16);
        nh[0] = WMMA_BF16(a0, b, nh[0]);
        nh[1] = WMMA_BF16(a1, b, nh[1]);
      }
      // GRU pointwise in accumulator registers
#pragma unroll
      for (int m = 0; m < 2; ++m)
#pragma unroll
        for (int v = 0; v < 8; ++v) {
          int row = m * 16 + v + ((lane & 16) ? 8 : 0);
          int col = cb + (lane & 15);
          float rr = 1.f / (1.f + expf(-(ar[m][v] + biasR[col])));
          float zz = 1.f / (1.f + expf(-(az[m][v] + biasZ[col])));
          float hp = (float)Abuf[a_off(row, 256 + col)];
          float na = tanhf(ni[m][v] + biasNi[col] + rr * (nh[m][v] + biasNh[col]));
          float hh = (1.f - zz) * na + zz * hp;
          Hbuf[a_off(row, col)] = (bf16)hh;
        }
    }
    __syncthreads();

    // ---- hc = [e | h] @ [res_W | comp_W]^T + (comp_b + res_b) ----
    v8f hc[2][4] = {};
#pragma unroll 4
    for (int kt = 0; kt < 40; ++kt) {
      const bf16* asrc = (kt < 8) ? (Abuf + kt * 1024) : (Hbuf + (kt - 8) * 1024);
      v16bf a0 = *(const v16bf*)(asrc + lane * 16);
      v16bf a1 = *(const v16bf*)(asrc + 512 + lane * 16);
#pragma unroll
      for (int nt = 0; nt < 4; ++nt) {
        v16bf b = *(const v16bf*)(Wc + ((size_t)((w * 4 + nt) * 40 + kt) * 32 + lane) * 16);
        hc[0][nt] = WMMA_BF16(a0, b, hc[0][nt]);
        hc[1][nt] = WMMA_BF16(a1, b, hc[1][nt]);
      }
    }

    // ---- LayerNorm reduction (lane-halves hold diff rows, same cols) ----
#pragma unroll
    for (int m = 0; m < 2; ++m)
#pragma unroll
      for (int v = 0; v < 8; ++v) {
        float ss = 0.f, qq = 0.f;
#pragma unroll
        for (int nt = 0; nt < 4; ++nt) {
          int col = w * 64 + nt * 16 + (lane & 15);
          float val = hc[m][nt][v] + biasC[col];
          hc[m][nt][v] = val;
          ss += val; qq += val * val;
        }
        for (int d = 8; d >= 1; d >>= 1) {
          ss += __shfl_xor(ss, d, 32);
          qq += __shfl_xor(qq, d, 32);
        }
        if ((lane & 15) == 0) {
          int row = m * 16 + v + ((lane & 16) ? 8 : 0);
          redS[w * 32 + row] = ss;
          redQ[w * 32 + row] = qq;
        }
      }
    __syncthreads();
    if (tid < 32) {
      float S = 0.f, Q = 0.f;
      for (int ww = 0; ww < 8; ++ww) { S += redS[ww * 32 + tid]; Q += redQ[ww * 32 + tid]; }
      float mu  = S * (1.0f / 512.0f);
      float var = Q * (1.0f / 512.0f) - mu * mu;
      stat[tid * 2]     = mu;
      stat[tid * 2 + 1] = rsqrtf(var + 1e-5f);
    }
    __syncthreads();

    // ---- normalize, emit fp32 output, seed h_above for next row ----
#pragma unroll
    for (int m = 0; m < 2; ++m)
#pragma unroll
      for (int nt = 0; nt < 4; ++nt)
#pragma unroll
        for (int v = 0; v < 8; ++v) {
          int row = m * 16 + v + ((lane & 16) ? 8 : 0);
          int col = w * 64 + nt * 16 + (lane & 15);
          float val = (hc[m][nt][v] - stat[row * 2]) * stat[row * 2 + 1] * lng[col] + lnb[col];
          out[(((size_t)row * 32 + i) * 32 + j) * 512 + col] = val;
          Abuf[a_off(row, 256 + col)] = (bf16)val;         // h_above for row i+1
          if (i == 31 && j == 31)
            out[(size_t)16777216 + (size_t)row * 512 + col] = val;  // h_final
        }
    __threadfence();
    __syncthreads();
    if (tid == 0)
      __hip_atomic_store(flags + (i * 32 + j), 1, __ATOMIC_RELEASE,
                         __HIP_MEMORY_SCOPE_AGENT);
  }
}

// ---------------------------------------------------------------------------
extern "C" void kernel_launch(void* const* d_in, const int* in_sizes, int n_in,
                              void* d_out, int out_size, void* d_ws, size_t ws_size,
                              hipStream_t stream) {
  const float* x      = (const float*)d_in[0];
  const float* emb_W  = (const float*)d_in[1];
  const float* emb_b  = (const float*)d_in[2];
  const float* W_ih   = (const float*)d_in[3];
  const float* b_ih   = (const float*)d_in[4];
  const float* W_hh   = (const float*)d_in[5];
  const float* b_hh   = (const float*)d_in[6];
  const float* comp_W = (const float*)d_in[7];
  const float* comp_b = (const float*)d_in[8];
  const float* res_W  = (const float*)d_in[9];
  const float* res_b  = (const float*)d_in[10];
  const float* ln_g   = (const float*)d_in[11];
  const float* ln_b   = (const float*)d_in[12];
  float* out = (float*)d_out;
  char*  ws  = (char*)d_ws;

  bf16* ePack = (bf16*)(ws + E_PACK_OFF);
  bf16* Wrz   = (bf16*)(ws + WRZ_OFF);
  bf16* Bni   = (bf16*)(ws + BNI_OFF);
  bf16* Bnh   = (bf16*)(ws + BNH_OFF);
  bf16* Wc    = (bf16*)(ws + WC_OFF);
  bf16* EmbB  = (bf16*)(ws + EMB_OFF);
  int*  flags = (int*)(ws + FLAG_OFF);

  hipMemsetAsync(flags, 0, 1024 * sizeof(int), stream);

  // Weight packing (fp32 -> bf16 WMMA B-fragment tiles)
  {
    int total = 128 * 40 * 512;  // Wrz: r,z rows of [W_ih | W_hh], N=2048 K=1280
    pack_b_kernel<<<(total + 255) / 256, 256, 0, stream>>>(
        Wrz, W_ih, 256, 0, W_hh, 1024, 0, 256, 40, total);
  }
  {
    int total = 64 * 8 * 512;    // Bni: n rows of W_ih, N=1024 K=256
    pack_b_kernel<<<(total + 255) / 256, 256, 0, stream>>>(
        Bni, W_ih, 256, 2048, W_ih, 256, 2048, 256, 8, total);
  }
  {
    int total = 64 * 32 * 512;   // Bnh: n rows of W_hh, N=1024 K=1024
    pack_b_kernel<<<(total + 255) / 256, 256, 0, stream>>>(
        Bnh, W_hh, 1024, 2048, W_hh, 1024, 2048, 1024, 32, total);
  }
  {
    int total = 32 * 40 * 512;   // Wc: [res_W | comp_W], N=512 K=1280
    pack_b_kernel<<<(total + 255) / 256, 256, 0, stream>>>(
        Wc, res_W, 256, 0, comp_W, 1024, 0, 256, 40, total);
  }
  {
    int total = 16 * 8 * 512;    // EmbB: emb_W, N=256 K=256
    pack_b_kernel<<<(total + 255) / 256, 256, 0, stream>>>(
        EmbB, emb_W, 256, 0, emb_W, 256, 0, 256, 8, total);
  }

  // Phase 1: e for all 1024 cells (fully parallel WMMA)
  embed_kernel<<<1024, 256, 0, stream>>>(x, emb_b, EmbB, ePack);

  // Phase 2: persistent wavefront, 32 workgroups (one per grid column)
  (void)hipFuncSetAttribute((const void*)wavefront_kernel,
                            hipFuncAttributeMaxDynamicSharedMemorySize,
                            WF_SMEM_BYTES);
  wavefront_kernel<<<32, 256, WF_SMEM_BYTES, stream>>>(
      ePack, Wrz, Bni, Bnh, Wc, b_ih, b_hh, comp_b, res_b, ln_g, ln_b,
      out, flags);

  (void)in_sizes; (void)n_in; (void)out_size; (void)ws_size;
}
